// RelationLevelAggregation_88055419503364
// MI455X (gfx1250) — compile-verified, matrix operand order checked by reading, not run on hardware
//
#include <hip/hip_runtime.h>

// Problem constants from the reference
#define NNODES 10000
#define KNBR   32
#define HDIM   512
#define NHEADS 8
#define NBINS  10
#define HEADDIM 64   // HDIM / NHEADS
#define NEG_SLOPE 0.2f

typedef _Float16 v8h  __attribute__((ext_vector_type(8)));
typedef _Float16 v16h __attribute__((ext_vector_type(16)));
typedef float    v8f  __attribute__((ext_vector_type(8)));

// ---------------------------------------------------------------------------
// f32 -> f16 conversion (z, P_w, W_w)
// ---------------------------------------------------------------------------
__global__ void f32_to_f16_kernel(const float* __restrict__ in,
                                  _Float16* __restrict__ out, int n) {
    int i = blockIdx.x * blockDim.x + threadIdx.x;
    if (i < n) out[i] = (_Float16)in[i];
}

// ---------------------------------------------------------------------------
// Fused GEMM: SUT[n, 0:512]    = z @ P_self^T
//             SUT[n, 512:1024] = z @ P_nbr^T
//             SUT[n, 1024:1536]= z @ W_w^T
// One wave computes a 64x64 output tile = 4x4 grid of 16x16 WMMA tiles,
// accumulating over H=512 in 16 K-steps of 32. 16 WMMAs per K-step reuse
// 4 A-subtiles x 4 B-subtiles -> 0.5 tile-loads per WMMA (4x less L2
// traffic than a 16x16-per-wave scheme).
// ---------------------------------------------------------------------------
__global__ __launch_bounds__(256)
void gemm_wmma_kernel(const _Float16* __restrict__ zh,
                      const _Float16* __restrict__ Pwh,   // 512 x 1024 (row major)
                      const _Float16* __restrict__ Wwh,   // 512 x 512
                      float* __restrict__ SUT)            // 10000 x 1536
{
    const int wid  = threadIdx.x >> 5;
    const int lane = threadIdx.x & 31;
    const int tile = blockIdx.x * 8 + wid;      // 157 * 24 = 3768 wave-tiles exactly
    const int mg   = tile / 24;                 // 64-row group (last one partial)
    const int ng   = tile - mg * 24;            // 64-col group
    const int half = lane >> 4;
    const int l16  = lane & 15;
    const int m0   = mg * 64;
    const int n0   = ng * 64;

    // Per-lane row pointers for the 4 A-subtiles and 4 B-subtiles.
    const _Float16* arow[4];
    const _Float16* brow[4];
    #pragma unroll
    for (int i = 0; i < 4; ++i) {
        int m = m0 + i * 16 + l16;
        if (m > NNODES - 1) m = NNODES - 1;     // clamp; stores are guarded below
        arow[i] = zh + (size_t)m * HDIM;
        const int n = n0 + i * 16 + l16;        // 64-wide tile never straddles a segment
        if (n < 512)       brow[i] = Pwh + (size_t)n * 1024;               // P_self row
        else if (n < 1024) brow[i] = Pwh + (size_t)(n - 512) * 1024 + 512; // P_nbr row
        else               brow[i] = Wwh + (size_t)(n - 1024) * 512;       // W_w row
    }

    v8f acc[4][4] = {};
    for (int kk = 0; kk < HDIM; kk += 32) {
        const int k0 = kk + half * 8;           // ISA K-chunk layout for 16-bit A/B
        v16h A[4], B[4];
        #pragma unroll
        for (int i = 0; i < 4; ++i) {
            v8h alo = *(const v8h*)(arow[i] + k0);
            v8h ahi = *(const v8h*)(arow[i] + k0 + 16);
            A[i] = __builtin_shufflevector(alo, ahi, 0,1,2,3,4,5,6,7,8,9,10,11,12,13,14,15);
            v8h blo = *(const v8h*)(brow[i] + k0);
            v8h bhi = *(const v8h*)(brow[i] + k0 + 16);
            B[i] = __builtin_shufflevector(blo, bhi, 0,1,2,3,4,5,6,7,8,9,10,11,12,13,14,15);
        }
        #pragma unroll
        for (int i = 0; i < 4; ++i)
            #pragma unroll
            for (int j = 0; j < 4; ++j)
                acc[i][j] = __builtin_amdgcn_wmma_f32_16x16x32_f16(
                    false, A[i], false, B[j], (short)0, acc[i][j], false, false);
    }

    // C/D layout: element v of lane L -> (M = v + 8*(L>>4), N = L&15)
    #pragma unroll
    for (int i = 0; i < 4; ++i) {
        #pragma unroll
        for (int v = 0; v < 8; ++v) {
            const int row = m0 + i * 16 + half * 8 + v;
            if (row < NNODES) {
                float* o = SUT + (size_t)row * 1536 + n0 + l16;
                #pragma unroll
                for (int j = 0; j < 4; ++j)
                    o[j * 16] = acc[i][j][v];
            }
        }
    }
}

// ---------------------------------------------------------------------------
// Fused edge kernel: one wave (32 lanes) per node; lane k owns neighbor k.
// Phase 1: scores + bin bias + mask + softmax (wave32 shfl reductions).
// Phase 2: lanes re-partitioned over the 512 output dims; attention-weighted
//          aggregation of gathered T rows + residual + leaky_relu + select.
// ---------------------------------------------------------------------------
__global__ __launch_bounds__(256)
void edge_kernel(const float* __restrict__ SUT,
                 const float* __restrict__ z,
                 const int*   __restrict__ nbr,
                 const int*   __restrict__ bins,
                 const float* __restrict__ y_w,     // 8 x 512
                 const float* __restrict__ c_bins,  // 10 x 8
                 const float* __restrict__ rw_p,
                 float* __restrict__ out)
{
    __shared__ float s_yw[NHEADS * HDIM];          // 16 KB
    __shared__ float s_attn[8][KNBR * NHEADS];     //  8 KB
    __shared__ int   s_idx[8][KNBR];               //  1 KB

    for (int i = threadIdx.x; i < NHEADS * HDIM; i += blockDim.x)
        s_yw[i] = y_w[i];
    __syncthreads();

    const int wid  = threadIdx.x >> 5;
    const int lane = threadIdx.x & 31;
    const int node = blockIdx.x * 8 + wid;         // 1250 * 8 = 10000 exactly

    // ---- Phase 1: per-neighbor scores ----
    const int  j_raw  = nbr[(size_t)node * KNBR + lane];
    const bool validk = (j_raw >= 0);
    const int  j      = validk ? j_raw : 0;
    const int  b      = validk ? bins[(size_t)node * KNBR + lane] : 0;

    float sc[NHEADS];
    #pragma unroll
    for (int e = 0; e < NHEADS; ++e) sc[e] = c_bins[b * NHEADS + e];

    const float4* S4 = (const float4*)(SUT + (size_t)node * 1536);
    const float4* U4 = (const float4*)(SUT + (size_t)j * 1536 + 512);
    for (int d4 = 0; d4 < HDIM / 4; ++d4) {
        float4 s = S4[d4];
        float4 u = U4[d4];
        float hv[4] = { s.x + u.x, s.y + u.y, s.z + u.z, s.w + u.w };
        #pragma unroll
        for (int t = 0; t < 4; ++t) {
            float h = hv[t] >= 0.f ? hv[t] : NEG_SLOPE * hv[t];
            const int d = d4 * 4 + t;
            #pragma unroll
            for (int e = 0; e < NHEADS; ++e)
                sc[e] = fmaf(h, s_yw[e * HDIM + d], sc[e]);
        }
    }
    #pragma unroll
    for (int e = 0; e < NHEADS; ++e)
        if (!validk) sc[e] = -1.0e9f;

    // softmax across the 32 lanes (k axis), per head
    #pragma unroll
    for (int e = 0; e < NHEADS; ++e) {
        float m = sc[e];
        for (int o = 16; o > 0; o >>= 1) m = fmaxf(m, __shfl_xor(m, o, 32));
        float ex = __expf(sc[e] - m);
        float sm = ex;
        for (int o = 16; o > 0; o >>= 1) sm += __shfl_xor(sm, o, 32);
        s_attn[wid][lane * NHEADS + e] = ex / sm;
    }
    s_idx[wid][lane] = j;
    const bool anyv = (__ballot(validk) != 0ULL);
    __syncthreads();

    // ---- Phase 2: aggregation; each lane owns 16 contiguous dims (one head) ----
    const int d0   = lane * 16;
    const int head = d0 >> 6;                     // each lane's 16 dims lie in one head
    float agg[16];
    #pragma unroll
    for (int t = 0; t < 16; ++t) agg[t] = 0.f;

    for (int k = 0; k < KNBR; ++k) {
        const int   jk = s_idx[wid][k];
        const float a  = s_attn[wid][k * NHEADS + head];
        const float4* T4 = (const float4*)(SUT + (size_t)jk * 1536 + 1024 + d0);
        #pragma unroll
        for (int q = 0; q < 4; ++q) {
            float4 t4 = T4[q];
            agg[q * 4 + 0] = fmaf(a, t4.x, agg[q * 4 + 0]);
            agg[q * 4 + 1] = fmaf(a, t4.y, agg[q * 4 + 1]);
            agg[q * 4 + 2] = fmaf(a, t4.z, agg[q * 4 + 2]);
            agg[q * 4 + 3] = fmaf(a, t4.w, agg[q * 4 + 3]);
        }
    }

    const float rw = rw_p[0];
    const float* zr   = z   + (size_t)node * HDIM + d0;
    float*       outr = out + (size_t)node * HDIM + d0;
    #pragma unroll
    for (int t = 0; t < 16; ++t) {
        float zi = zr[t];
        float o  = fmaf(rw, zi, agg[t]);
        o = o >= 0.f ? o : NEG_SLOPE * o;
        outr[t] = anyv ? o : zi;
    }
}

// ---------------------------------------------------------------------------
extern "C" void kernel_launch(void* const* d_in, const int* in_sizes, int n_in,
                              void* d_out, int out_size, void* d_ws, size_t ws_size,
                              hipStream_t stream) {
    const float* z      = (const float*)d_in[0];
    // d_in[1] = A (N x N zeros) -- unused by the computation
    const int*   nbr    = (const int*)d_in[2];
    const int*   bins   = (const int*)d_in[3];
    const float* P_w    = (const float*)d_in[4];
    const float* y_w    = (const float*)d_in[5];
    const float* W_w    = (const float*)d_in[6];
    const float* c_bins = (const float*)d_in[7];
    const float* rw     = (const float*)d_in[8];
    float* out = (float*)d_out;

    // Workspace layout (all offsets 16B-aligned):
    //   zh  : 10000*512  f16 = 10,240,000 B
    //   Pwh :   512*1024 f16 =  1,048,576 B
    //   Wwh :   512*512  f16 =    524,288 B
    //   SUT : 10000*1536 f32 = 61,440,000 B   (total ~73.3 MB)
    char* ws = (char*)d_ws;
    _Float16* zh  = (_Float16*)(ws);
    _Float16* Pwh = (_Float16*)(ws + 10240000);
    _Float16* Wwh = (_Float16*)(ws + 11288576);
    float*    SUT = (float*)   (ws + 11812864);

    f32_to_f16_kernel<<<(NNODES * HDIM + 255) / 256, 256, 0, stream>>>(z,   zh,  NNODES * HDIM);
    f32_to_f16_kernel<<<(512 * 1024   + 255) / 256, 256, 0, stream>>>(P_w, Pwh, 512 * 1024);
    f32_to_f16_kernel<<<(512 * 512    + 255) / 256, 256, 0, stream>>>(W_w, Wwh, 512 * 512);

    // 157 M-groups * 24 N-groups = 3768 wave-tiles; 8 waves/block -> 471 blocks
    gemm_wmma_kernel<<<471, 256, 0, stream>>>(zh, Pwh, Wwh, SUT);

    // 8 nodes (waves) per block -> 1250 blocks
    edge_kernel<<<1250, 256, 0, stream>>>(SUT, z, nbr, bins, y_w, c_bins, rw, out);
}
